// RPN_48301202211090
// MI455X (gfx1250) — matrix operand order repaired and checked by reading, not compiled, not run on hardware
//
#include <hip/hip_runtime.h>

// ---------------- problem constants ----------------
#define BATCH 8
#define CIN   512
#define COUT  512
#define HH    50
#define WW    50
#define NPIX  (HH*WW)        // 2500
#define PH    52             // padded spatial
#define NANCH 22500          // 50*50*9
#define PRE_NMS 1000
#define POST_NMS 300
#define SORTN 32768
#define NKSTEP 144           // 9 positions * 16 ci-chunks of 32

typedef __attribute__((ext_vector_type(16))) _Float16 v16h;
typedef __attribute__((ext_vector_type(8)))  float    v8f;
typedef __attribute__((ext_vector_type(4)))  int      v4i;

// -------- CDNA5 async global->LDS path (guarded; sync fallback otherwise) --------
#if defined(__HIP_DEVICE_COMPILE__) && __has_builtin(__builtin_amdgcn_global_load_async_to_lds_b128)
#define USE_ASYNC_LDS 1
#else
#define USE_ASYNC_LDS 0
#endif

__device__ __forceinline__ void copy16_to_lds(void* lds_dst, const void* gsrc) {
#if USE_ASYNC_LDS
  __builtin_amdgcn_global_load_async_to_lds_b128((v4i*)gsrc, (v4i*)lds_dst, 0, 0);
#else
  *(uint4*)lds_dst = *(const uint4*)gsrc;
#endif
}

__device__ __forceinline__ void wait_async_lds() {
#if USE_ASYNC_LDS
#if __has_builtin(__builtin_amdgcn_s_wait_asynccnt)
  __builtin_amdgcn_s_wait_asynccnt(0);
#else
  asm volatile("s_wait_asynccnt 0x0" ::: "memory");
#endif
#endif
}

// ---------------- 1) pad + convert x: NCHW f32 -> NHWC f16 padded ----------------
__global__ void pad_convert_x(const float* __restrict__ x, _Float16* __restrict__ xpad) {
  long i = (long)blockIdx.x * blockDim.x + threadIdx.x;   // over B*52*52*512
  if (i >= (long)BATCH * PH * PH * CIN) return;
  int c  = (int)(i % CIN); long t = i / CIN;
  int pw = (int)(t % PH);  t /= PH;
  int ph = (int)(t % PH);  int b = (int)(t / PH);
  float v = 0.0f;
  int h = ph - 1, w = pw - 1;
  if (h >= 0 && h < HH && w >= 0 && w < WW)
    v = x[(((long)b * CIN + c) * HH + h) * WW + w];
  xpad[i] = (_Float16)v;
}

// ---------------- 2) convert weights: [co][ci][3][3] f32 -> [pos][co][ci] f16 ----------------
__global__ void convert_w(const float* __restrict__ ws, _Float16* __restrict__ w2) {
  long i = (long)blockIdx.x * blockDim.x + threadIdx.x;   // 9*512*512
  if (i >= (long)9 * COUT * CIN) return;
  int ci = (int)(i % CIN); long t = i / CIN;
  int co = (int)(t % COUT); int pos = (int)(t / COUT);
  w2[i] = (_Float16)ws[((long)co * CIN + ci) * 9 + pos];
}

// ---------------- 3) 3x3 conv as implicit GEMM via WMMA f16 ----------------
// 64 pixels x 128 co per workgroup, double-buffered LDS, async stage overlap.
__global__ __launch_bounds__(256)
void conv3_wmma(const _Float16* __restrict__ xpad,   // [B][52][52][512]
                const _Float16* __restrict__ w2,     // [9][512co][512ci]
                const float*    __restrict__ bias,   // [512]
                float*          __restrict__ out) {  // [B][512][2500]
  __shared__ _Float16 lA[2][64][32];    // 2 x 4KB
  __shared__ _Float16 lB[2][128][32];   // 2 x 8KB

  const int b  = blockIdx.z;
  const int mt = blockIdx.x;                 // 0..39 (pixel tiles of 64)
  const int nt = blockIdx.y;                 // 0..3  (co tiles of 128)
  const int tid   = threadIdx.x;
  const int wave  = tid >> 5;                // 0..7
  const int lane  = tid & 31;
  const int laneM = lane & 15;
  const int hi    = lane >> 4;               // 0/1

  const int mw    = wave & 3;                // M subtile 0..3 (16 pixels)
  const int nbase = (wave >> 2) * 4;         // N subtiles {nbase..nbase+3} of 8

  // staging assignment: thread -> (row p, 16B quarter q)
  const int p = tid >> 2;                    // 0..63
  const int q = tid & 3;                     // 0..3

  int gp = mt * 64 + p;
  if (gp >= NPIX) gp = NPIX - 1;             // clamp; dead outputs masked on store
  const int h = gp / WW, w = gp % WW;

  v8f acc[4] = {{}, {}, {}, {}};

  union Frag { uint4 u[2]; v16h v; };

  auto stage = [&](int it, int buf) {
    const int pos = it >> 4;                 // /16
    const int c0  = (it & 15) << 5;          // *32
    const int kh = pos / 3, kw = pos % 3;
    const _Float16* asrc  = xpad + (((long)b * PH + (h + kh)) * PH + (w + kw)) * CIN + c0 + q * 8;
    const _Float16* bsrc0 = w2 + ((long)pos * COUT + nt * 128 + p) * CIN + c0 + q * 8;
    copy16_to_lds(&lA[buf][p][q * 8], asrc);
    copy16_to_lds(&lB[buf][p][q * 8], bsrc0);
    copy16_to_lds(&lB[buf][p + 64][q * 8], bsrc0 + (long)64 * CIN);
  };

  stage(0, 0);
  for (int it = 0; it < NKSTEP; ++it) {
    const int cur = it & 1;
    wait_async_lds();                        // my async stage for buf `cur` done
    __syncthreads();                         // everyone's stage visible; buf cur^1 free
    if (it + 1 < NKSTEP) stage(it + 1, cur ^ 1);

    // A fragment: lane holds M=laneM, K = [hi*16, hi*16+16) contiguous
    Frag fa;
    fa.u[0] = *(const uint4*)&lA[cur][mw * 16 + laneM][hi * 16];
    fa.u[1] = *(const uint4*)&lA[cur][mw * 16 + laneM][hi * 16 + 8];
    // B fragments: lane holds N=laneM, K = [hi*16, hi*16+16) contiguous
    Frag fb[4];
    #pragma unroll
    for (int i4 = 0; i4 < 4; ++i4) {
      fb[i4].u[0] = *(const uint4*)&lB[cur][(nbase + i4) * 16 + laneM][hi * 16];
      fb[i4].u[1] = *(const uint4*)&lB[cur][(nbase + i4) * 16 + laneM][hi * 16 + 8];
    }
    #pragma unroll
    for (int i4 = 0; i4 < 4; ++i4)
      acc[i4] = __builtin_amdgcn_wmma_f32_16x16x32_f16(false, fa.v, false, fb[i4].v,
                                                       (short)0, acc[i4], false, false);
  }

  // store D: VGPR r -> pixel = pixbase + r, lane -> co
  const int pixbase = mt * 64 + mw * 16 + hi * 8;
  #pragma unroll
  for (int i4 = 0; i4 < 4; ++i4) {
    const int co = nt * 128 + (nbase + i4) * 16 + laneM;
    const float bs = bias[co];
    float* op = out + ((long)b * COUT + co) * NPIX + pixbase;
    #pragma unroll
    for (int r = 0; r < 8; ++r)
      if (pixbase + r < NPIX) op[r] = fmaxf(acc[i4][r] + bs, 0.0f);
  }
}

// ---------------- 4) 1x1 convs (cls 18, reg 36) ----------------
__global__ void conv1x1(const float* __restrict__ sh,
                        const float* __restrict__ wc, const float* __restrict__ bc,
                        const float* __restrict__ wr, const float* __restrict__ br,
                        float* __restrict__ clso, float* __restrict__ rego) {
  long i = (long)blockIdx.x * blockDim.x + threadIdx.x;  // B*2500*54
  if (i >= (long)BATCH * NPIX * 54) return;
  int oc = (int)(i % 54); long t = i / 54;
  int pix = (int)(t % NPIX); int b = (int)(t / NPIX);
  const float* wp = (oc < 18) ? (wc + (long)oc * CIN) : (wr + (long)(oc - 18) * CIN);
  float s = (oc < 18) ? bc[oc] : br[oc - 18];
  const float* sp = sh + (long)b * COUT * NPIX + pix;
  for (int c = 0; c < CIN; ++c) s += sp[(long)c * NPIX] * wp[c];
  if (oc < 18) clso[((long)b * NPIX + pix) * 18 + oc]        = s;
  else         rego[((long)b * NPIX + pix) * 36 + (oc - 18)] = s;
}

// ---------------- 5) softmax + bbox decode + clip + min-size filter ----------------
__global__ void decode(const float* __restrict__ clso, const float* __restrict__ rego,
                       float* __restrict__ boxes, float* __restrict__ score) {
  long i = (long)blockIdx.x * blockDim.x + threadIdx.x;  // B*22500
  if (i >= (long)BATCH * NANCH) return;
  int n = (int)(i % NANCH); int b = (int)(i / NANCH);
  int pix = n / 9, a = n % 9;
  int r = a / 3, s = a % 3;
  const float ratios[3] = {0.5f, 1.0f, 2.0f};
  const float scales[3] = {8.0f, 16.0f, 32.0f};
  float ah = 16.0f * scales[s] * sqrtf(ratios[r]);
  float aw = 16.0f * scales[s] * sqrtf(1.0f / ratios[r]);
  int hh = pix / WW, ww = pix % WW;
  float cy = hh * 16.0f + 8.0f, cx = ww * 16.0f + 8.0f;

  const float* lg = clso + ((long)b * NPIX + pix) * 18 + a * 2;
  float m  = fmaxf(lg[0], lg[1]);
  float e0 = __expf(lg[0] - m), e1 = __expf(lg[1] - m);
  float p1 = e1 / (e0 + e1);

  const float* lc = rego + ((long)b * NPIX + pix) * 36 + a * 4;
  float ncy = lc[0] * ah + cy;
  float ncx = lc[1] * aw + cx;
  float nh  = ah * __expf(lc[2]);
  float nw  = aw * __expf(lc[3]);
  float y1 = fminf(fmaxf(ncy - 0.5f * nh, 0.0f), 800.0f);
  float x1 = fminf(fmaxf(ncx - 0.5f * nw, 0.0f), 800.0f);
  float y2 = fminf(fmaxf(ncy + 0.5f * nh, 0.0f), 800.0f);
  float x2 = fminf(fmaxf(ncx + 0.5f * nw, 0.0f), 800.0f);
  float sc = ((y2 - y1) >= 16.0f && (x2 - x1) >= 16.0f) ? p1 : -__builtin_inff();
  float* bp = boxes + i * 4;
  bp[0] = y1; bp[1] = x1; bp[2] = y2; bp[3] = x2;
  score[i] = sc;
}

// ---------------- 6) per-batch top-1000 via in-LDS bitonic sort (256KB LDS) ----------------
extern __shared__ unsigned char smem_raw[];
__global__ __launch_bounds__(1024)
void sortk(const float* __restrict__ score, const float* __restrict__ boxes,
           float* __restrict__ sboxes, float* __restrict__ sscore) {
  float* ss = (float*)smem_raw;               // 32768 floats
  int*   si = (int*)(smem_raw + SORTN * 4);   // 32768 ints
  const int b = blockIdx.x;
  for (int i = threadIdx.x; i < SORTN; i += 1024) {
    ss[i] = (i < NANCH) ? score[(long)b * NANCH + i] : -__builtin_inff();
    si[i] = i;
  }
  __syncthreads();
  for (int k = 2; k <= SORTN; k <<= 1) {
    for (int j = k >> 1; j > 0; j >>= 1) {
      for (int i = threadIdx.x; i < SORTN; i += 1024) {
        int ix = i ^ j;
        if (ix > i) {
          bool desc = ((i & k) == 0);
          float va = ss[i], vb = ss[ix];
          if (desc ? (va < vb) : (va > vb)) {
            ss[i] = vb; ss[ix] = va;
            int t = si[i]; si[i] = si[ix]; si[ix] = t;
          }
        }
      }
      __syncthreads();
    }
  }
  for (int i = threadIdx.x; i < PRE_NMS; i += 1024) {
    int idx = si[i];
    sscore[(long)b * PRE_NMS + i] = ss[i];
    float4 v4 = make_float4(0.f, 0.f, 0.f, 0.f);
    if (idx < NANCH) v4 = *(const float4*)(boxes + ((long)b * NANCH + idx) * 4);
    *(float4*)(sboxes + ((long)b * PRE_NMS + i) * 4) = v4;
  }
}

// ---------------- 7) greedy NMS + top-300 compaction ----------------
__global__ __launch_bounds__(1024)
void nms(const float* __restrict__ sboxes, const float* __restrict__ sscore,
         float* __restrict__ out) {
  __shared__ float by1[PRE_NMS], bx1[PRE_NMS], by2[PRE_NMS], bx2[PRE_NMS];
  __shared__ float sc[PRE_NMS];
  __shared__ int   kept[PRE_NMS];
  __shared__ int   supp;
  const int b = blockIdx.x;
  for (int i = threadIdx.x; i < PRE_NMS; i += 1024) {
    const float* p = sboxes + ((long)b * PRE_NMS + i) * 4;
    by1[i] = p[0]; bx1[i] = p[1]; by2[i] = p[2]; bx2[i] = p[3];
    sc[i] = sscore[(long)b * PRE_NMS + i];
    kept[i] = 1;
  }
  __syncthreads();
  for (int i = 1; i < PRE_NMS; ++i) {
    if (threadIdx.x == 0) supp = 0;
    __syncthreads();
    float ay1 = by1[i], ax1 = bx1[i], ay2 = by2[i], ax2 = bx2[i];
    float aarea = (ay2 - ay1) * (ax2 - ax1);
    int local = 0;
    for (int j = threadIdx.x; j < i; j += 1024) {
      if (kept[j]) {
        float ty = fmaxf(ay1, by1[j]);
        float tx = fmaxf(ax1, bx1[j]);
        float cy = fminf(ay2, by2[j]);
        float cx = fminf(ax2, bx2[j]);
        float inter = fmaxf(cy - ty, 0.0f) * fmaxf(cx - tx, 0.0f);
        float ja = (by2[j] - by1[j]) * (bx2[j] - bx1[j]);
        float iou = inter / (aarea + ja - inter + 1e-9f);
        if (iou > 0.7f) local = 1;
      }
    }
    if (local) atomicOr(&supp, 1);
    __syncthreads();
    if (threadIdx.x == 0 && supp) kept[i] = 0;
    __syncthreads();
  }
  if (threadIdx.x == 0) {
    int cnt = 0;
    for (int n = 0; n < PRE_NMS && cnt < POST_NMS; ++n)
      if (kept[n] && sc[n] > -1e29f) {
        float* o = out + ((long)b * POST_NMS + cnt) * 5;
        o[0] = by1[n]; o[1] = bx1[n]; o[2] = by2[n]; o[3] = bx2[n]; o[4] = sc[n];
        ++cnt;
      }
    for (int n = 0; n < PRE_NMS && cnt < POST_NMS; ++n)
      if (!(kept[n] && sc[n] > -1e29f)) {
        float* o = out + ((long)b * POST_NMS + cnt) * 5;
        o[0] = by1[n]; o[1] = bx1[n]; o[2] = by2[n]; o[3] = bx2[n]; o[4] = -1e30f;
        ++cnt;
      }
  }
}

// ---------------- host ----------------
extern "C" void kernel_launch(void* const* d_in, const int* in_sizes, int n_in,
                              void* d_out, int out_size, void* d_ws, size_t ws_size,
                              hipStream_t stream) {
  const float* x       = (const float*)d_in[0];
  const float* w_share = (const float*)d_in[1];
  const float* b_share = (const float*)d_in[2];
  const float* w_cls   = (const float*)d_in[3];
  const float* b_cls   = (const float*)d_in[4];
  const float* w_reg   = (const float*)d_in[5];
  const float* b_reg   = (const float*)d_in[6];
  float* out = (float*)d_out;

  unsigned char* ws = (unsigned char*)d_ws;
  size_t off = 0;
  _Float16* xpad = (_Float16*)(ws + off); off += (size_t)BATCH * PH * PH * CIN * 2;   // 22.15 MB
  _Float16* w2   = (_Float16*)(ws + off); off += (size_t)9 * COUT * CIN * 2;          //  4.72 MB
  float* shared  = (float*)(ws + off);    off += (size_t)BATCH * COUT * NPIX * 4;     // 40.96 MB
  float* clso    = (float*)(ws + off);    off += (size_t)BATCH * NPIX * 18 * 4;
  float* rego    = (float*)(ws + off);    off += (size_t)BATCH * NPIX * 36 * 4;
  float* boxes   = (float*)(ws + off);    off += (size_t)BATCH * NANCH * 4 * 4;
  float* score   = (float*)(ws + off);    off += (size_t)BATCH * NANCH * 4;
  float* sboxes  = (float*)(ws + off);    off += (size_t)BATCH * PRE_NMS * 4 * 4;
  float* sscore  = (float*)(ws + off);    off += (size_t)BATCH * PRE_NMS * 4;
  (void)ws_size; (void)in_sizes; (void)n_in; (void)out_size;

  {
    long n = (long)BATCH * PH * PH * CIN;
    pad_convert_x<<<dim3((unsigned)((n + 255) / 256)), dim3(256), 0, stream>>>(x, xpad);
  }
  {
    long n = (long)9 * COUT * CIN;
    convert_w<<<dim3((unsigned)((n + 255) / 256)), dim3(256), 0, stream>>>(w_share, w2);
  }
  conv3_wmma<<<dim3(40, 4, BATCH), dim3(256), 0, stream>>>(xpad, w2, b_share, shared);
  {
    long n = (long)BATCH * NPIX * 54;
    conv1x1<<<dim3((unsigned)((n + 255) / 256)), dim3(256), 0, stream>>>(
        shared, w_cls, b_cls, w_reg, b_reg, clso, rego);
  }
  {
    long n = (long)BATCH * NANCH;
    decode<<<dim3((unsigned)((n + 255) / 256)), dim3(256), 0, stream>>>(clso, rego, boxes, score);
  }
  sortk<<<dim3(BATCH), dim3(1024), SORTN * 8, stream>>>(score, boxes, sboxes, sscore);
  nms<<<dim3(BATCH), dim3(1024), 0, stream>>>(sboxes, sscore, out);
}
